// FractalAttention_15710990369336
// MI455X (gfx1250) — matrix operand order repaired and checked by reading, not compiled, not run on hardware
//
#include <hip/hip_runtime.h>

typedef __attribute__((ext_vector_type(16))) __bf16 v16bf;
typedef __attribute__((ext_vector_type(8)))  __bf16 v8bf;
typedef __attribute__((ext_vector_type(8)))  float  v8f;

#define HWSZ   65536     // 256*256
#define CHN    64
#define BATCH  8
#define NTOT   524288.0f // 8*65536 samples per channel for BN
#define BN_EPS 1e-5f
#define SMOOTH 1e-5f

__device__ __forceinline__ float sigm(float x) {
    return 1.0f / (1.0f + __expf(-x));
}

// fractal tanimoto: (tpl+eps) * sum_{d<5} 1/(2^d*sqs - (2^{d+1}-1)*tpl + eps) / 5
__device__ __forceinline__ float fractal5(float tpl, float sqs) {
    float den = 0.f, a = 1.f;
#pragma unroll
    for (int d = 0; d < 5; ++d) {
        float b = -(2.f * a - 1.f);
        den += 1.f / (a * sqs + b * tpl + SMOOTH);
        a *= 2.f;
    }
    return (tpl + SMOOTH) * den * 0.2f;
}

// pair over (y,yhat) and complements, Nc = size of reduced dim
__device__ __forceinline__ float fractal_pair(float tpl, float qq, float kk,
                                              float sq, float sk, float Nc) {
    float c1   = fractal5(tpl, qq + kk);
    float tpl2 = Nc - sq - sk + tpl;
    float sqs2 = 2.f * Nc - 2.f * (sq + sk) + qq + kk;
    float c2   = fractal5(tpl2, sqs2);
    return 0.5f * (c1 + c2);
}

// ---------------------------------------------------------------------------
// K1: pack OIHW fp32 weights into WMMA B-fragment order (bf16).
// Fragment id = ((conv*4+grp)*18 + kc)*32 + lane, 16 halves per lane.
// B layout (32x16 KxN, 16-bit): n = lane%16, K = (half<8?0:16) + (lane>=16 ? 8 : 0) + (half&7).
// kc = (kh*3+kw)*2 + ci_half;  ci = ci_half*32 + K.
// ---------------------------------------------------------------------------
__global__ void pack_weights(const float* __restrict__ wq,
                             const float* __restrict__ wk,
                             const float* __restrict__ wv,
                             __bf16* __restrict__ pB) {
    int idx = blockIdx.x * blockDim.x + threadIdx.x;
    if (idx >= 110592) return;           // 3*4*18*32*16
    int hh   = idx & 15;
    int lane = (idx >> 4) & 31;
    int kc   = (idx >> 9) % 18;
    int g2   = (idx >> 9) / 18;          // conv*4 + grp
    int grp  = g2 & 3;
    int conv = g2 >> 2;
    int co   = (grp << 4) + (lane & 15);
    int K    = (hh & 7) + ((lane >> 4) << 3) + ((hh >> 3) << 4);
    int chh  = kc & 1;
    int t    = kc >> 1;
    int kh   = t / 3, kw = t - kh * 3;
    int ci   = (chh << 5) + K;
    const float* w = (conv == 0) ? wq : (conv == 1) ? wk : wv;
    pB[idx] = (__bf16)w[((co * 64 + ci) * 3 + kh) * 3 + kw];
}

// ---------------------------------------------------------------------------
// K2: implicit-GEMM 3x3 conv for q,k,v via v_wmma_f32_16x16x32_bf16.
// Block = 128 thr (4 waves). Tile = 64 pixels (one row) x 64 out-ch x 3 convs.
// Each wave: co group = waveId, 4 M-tiles of 16 pixels -> B fragments reused
// 4x, 12 WMMAs per K-chunk against 6 global_b128 + 8 ds_b128.
// All 3 halo rows staged in LDS up front: single barrier, then 216 WMMAs.
// A layout (16x32 MxK, 16-bit): m = lane%16,
//   K = (h<8?0:16) + (lane>=16?8:0) + (h&7)  -> two contiguous 16B LDS reads.
// ---------------------------------------------------------------------------
__global__ __launch_bounds__(128) void conv_qkv_wmma(
        const float* __restrict__ x, const __bf16* __restrict__ pB,
        __bf16* __restrict__ yq, __bf16* __restrict__ yk,
        __bf16* __restrict__ yv) {
    __shared__ __bf16 xt[3 * 66 * 72];   // [row 0..2][px 0..65][ci 0..63]

    const int tid  = threadIdx.x;
    const int lane = tid & 31;
    const int wvid = tid >> 5;           // co group 0..3
    const int n  = blockIdx.z;
    const int h  = blockIdx.y;
    const int w0 = blockIdx.x << 6;      // 64-pixel tile

    const int m   = lane & 15;           // pixel within M tile
    const int lg8 = (lane >> 4) << 3;    // 0 or 8 (A K-offset / store offset)

    // ---- stage x halo slab: rows h-1..h+1, cols w0-1..w0+64, all 64 ci ----
    for (int e = tid; e < 3 * 66 * 64; e += 128) {
        int row = e / (66 * 64);
        int r2  = e - row * (66 * 64);
        int ci  = r2 / 66;
        int px  = r2 - ci * 66;
        int hy  = h + row - 1;
        int wx  = w0 + px - 1;
        float val = 0.f;
        if (hy >= 0 && hy < 256 && wx >= 0 && wx < 256)
            val = x[((n * 64 + ci) * 256 + hy) * 256 + wx];
        xt[(row * 66 + px) * 72 + ci] = (__bf16)val;
    }
    __syncthreads();

    v8f accq[4] = {{}, {}, {}, {}};
    v8f acck[4] = {{}, {}, {}, {}};
    v8f accv[4] = {{}, {}, {}, {}};

#pragma unroll
    for (int kh = 0; kh < 3; ++kh) {
#pragma unroll
        for (int kw = 0; kw < 3; ++kw) {
#pragma unroll
            for (int chh = 0; chh < 2; ++chh) {
                int kc = (kh * 3 + kw) * 2 + chh;
                const __bf16* bp = pB + (((wvid * 18 + kc) * 32 + lane) << 4);
                v16bf bq = *(const v16bf*)bp;
                v16bf bk = *(const v16bf*)(bp + 36864);   // +1 conv stride
                v16bf bv = *(const v16bf*)(bp + 73728);
                int cb = (chh << 5) + lg8;
#pragma unroll
                for (int mt = 0; mt < 4; ++mt) {
                    int px = (mt << 4) + m + kw;
                    const __bf16* ap = &xt[(kh * 66 + px) * 72 + cb];
                    v8bf lo = *(const v8bf*)ap;          // ci cb..cb+7
                    v8bf hi = *(const v8bf*)(ap + 16);   // ci cb+16..cb+23
                    v16bf a = __builtin_shufflevector(lo, hi,
                            0,1,2,3,4,5,6,7,8,9,10,11,12,13,14,15);
                    accq[mt] = __builtin_amdgcn_wmma_f32_16x16x32_bf16(
                            false, a, false, bq, (short)0, accq[mt], false, false);
                    acck[mt] = __builtin_amdgcn_wmma_f32_16x16x32_bf16(
                            false, a, false, bk, (short)0, acck[mt], false, false);
                    accv[mt] = __builtin_amdgcn_wmma_f32_16x16x32_bf16(
                            false, a, false, bv, (short)0, accv[mt], false, false);
                }
            }
        }
    }

    // C layout: lane holds col N=lane%16 (=co), rows M = r + 8*(lane>=16)
    // -> each lane owns 8 CONSECUTIVE pixels per tile: one b128 store each.
    const int co   = (wvid << 4) + m;
    const int base = (n * 64 + co) * HWSZ + h * 256 + w0 + lg8;
#pragma unroll
    for (int mt = 0; mt < 4; ++mt) {
        v8bf oq, ok, ov;
#pragma unroll
        for (int r = 0; r < 8; ++r) {
            oq[r] = (__bf16)accq[mt][r];
            ok[r] = (__bf16)acck[mt][r];
            ov[r] = (__bf16)accv[mt][r];
        }
        *(v8bf*)(yq + base + (mt << 4)) = oq;
        *(v8bf*)(yk + base + (mt << 4)) = ok;
        *(v8bf*)(yv + base + (mt << 4)) = ov;
    }
}

// ---------------------------------------------------------------------------
// K3: BN1 stats — one block per (conv, channel); deterministic tree reduce.
// ---------------------------------------------------------------------------
__global__ __launch_bounds__(256) void bn1_stats(
        const __bf16* __restrict__ yq, const __bf16* __restrict__ yk,
        const __bf16* __restrict__ yv, float* __restrict__ stats) {
    int c = blockIdx.x, conv = blockIdx.y;
    const __bf16* y = (conv == 0) ? yq : (conv == 1) ? yk : yv;
    float s = 0.f, s2 = 0.f;
    for (int n = 0; n < BATCH; ++n) {
        int base = (n * 64 + c) * HWSZ;
        for (int i = threadIdx.x; i < HWSZ; i += 256) {
            float v = (float)y[base + i];
            s += v; s2 += v * v;
        }
    }
    __shared__ float r0[256], r1[256];
    r0[threadIdx.x] = s; r1[threadIdx.x] = s2;
    __syncthreads();
    for (int o = 128; o > 0; o >>= 1) {
        if (threadIdx.x < o) {
            r0[threadIdx.x] += r0[threadIdx.x + o];
            r1[threadIdx.x] += r1[threadIdx.x + o];
        }
        __syncthreads();
    }
    if (threadIdx.x == 0) {
        stats[(conv * 64 + c) * 2 + 0] = r0[0];
        stats[(conv * 64 + c) * 2 + 1] = r1[0];
    }
}

__global__ void finalize_bn1(const float* __restrict__ stats,
                             const float* gq, const float* bq,
                             const float* gk, const float* bk,
                             const float* gv, const float* bv,
                             float* __restrict__ ab) {
    int t = threadIdx.x;                 // 0..191
    int conv = t >> 6, c = t & 63;
    float mean = stats[t * 2] / NTOT;
    float var  = stats[t * 2 + 1] / NTOT - mean * mean;
    const float* g = (conv == 0) ? gq : (conv == 1) ? gk : gv;
    const float* b = (conv == 0) ? bq : (conv == 1) ? bk : bv;
    float sc = g[c] * rsqrtf(var + BN_EPS);
    ab[t * 2 + 0] = sc;
    ab[t * 2 + 1] = b[c] - mean * sc;
}

// ---------------------------------------------------------------------------
// K4: per-pixel channel reductions + spatial attention (dims=(1,), Nc=64).
// ---------------------------------------------------------------------------
__global__ __launch_bounds__(256) void pixel_attn(
        const __bf16* __restrict__ yq, const __bf16* __restrict__ yk,
        const float* __restrict__ ab, float* __restrict__ attn_s) {
    __shared__ float sab[256];           // conv0(q): 0..127, conv1(k): 128..255
    sab[threadIdx.x] = ab[threadIdx.x];
    __syncthreads();
    int p  = blockIdx.x * 256 + threadIdx.x;   // 0..524287
    int n  = p >> 16;
    int hw = p & 65535;
    int base = n * 64 * HWSZ + hw;
    float tpl = 0.f, qq = 0.f, kk = 0.f, sq = 0.f, sk = 0.f;
#pragma unroll 4
    for (int c = 0; c < 64; ++c) {
        float q = sigm(sab[c * 2] * (float)yq[base + c * HWSZ] + sab[c * 2 + 1]);
        float k = sigm(sab[128 + c * 2] * (float)yk[base + c * HWSZ] + sab[128 + c * 2 + 1]);
        tpl += q * k; qq += q * q; kk += k * k; sq += q; sk += k;
    }
    attn_s[p] = fractal_pair(tpl, qq, kk, sq, sk, 64.f);
}

// ---------------------------------------------------------------------------
// K5: per-(b,c) spatial reductions + channel attention (dims=(2,3), Nc=65536).
// ---------------------------------------------------------------------------
__global__ __launch_bounds__(256) void channel_attn(
        const __bf16* __restrict__ yq, const __bf16* __restrict__ yk,
        const float* __restrict__ ab, float* __restrict__ attn_c) {
    int c = blockIdx.x, n = blockIdx.y, t = threadIdx.x;
    float aq = ab[c * 2], bq = ab[c * 2 + 1];
    float ak = ab[128 + c * 2], bk = ab[128 + c * 2 + 1];
    int base = (n * 64 + c) * HWSZ;
    float tpl = 0.f, qq = 0.f, kk = 0.f, sq = 0.f, sk = 0.f;
    for (int i = t; i < HWSZ; i += 256) {
        float q = sigm(aq * (float)yq[base + i] + bq);
        float k = sigm(ak * (float)yk[base + i] + bk);
        tpl += q * k; qq += q * q; kk += k * k; sq += q; sk += k;
    }
    __shared__ float red[5][256];
    red[0][t] = tpl; red[1][t] = qq; red[2][t] = kk; red[3][t] = sq; red[4][t] = sk;
    __syncthreads();
    for (int o = 128; o > 0; o >>= 1) {
        if (t < o)
#pragma unroll
            for (int v = 0; v < 5; ++v) red[v][t] += red[v][t + o];
        __syncthreads();
    }
    if (t == 0)
        attn_c[n * 64 + c] = fractal_pair(red[0][0], red[1][0], red[2][0],
                                          red[3][0], red[4][0], 65536.f);
}

// ---------------------------------------------------------------------------
// K6: final-BN partial stats per (c, n) — deterministic, no atomics.
// ---------------------------------------------------------------------------
__global__ __launch_bounds__(256) void bn2_partials(
        const __bf16* __restrict__ yv, const float* __restrict__ ab,
        const float* __restrict__ attn_s, const float* __restrict__ attn_c,
        float* __restrict__ part) {
    int c = blockIdx.x, n = blockIdx.y, t = threadIdx.x;
    float av = ab[256 + c * 2], bv = ab[256 + c * 2 + 1];
    float ac = attn_c[n * 64 + c];
    int base = (n * 64 + c) * HWSZ, sbase = n * HWSZ;
    float s = 0.f, s2 = 0.f;
    for (int i = t; i < HWSZ; i += 256) {
        float v   = sigm(av * (float)yv[base + i] + bv);
        float att = 0.5f * (attn_s[sbase + i] + ac) * v;
        s += att; s2 += att * att;
    }
    __shared__ float r0[256], r1[256];
    r0[t] = s; r1[t] = s2;
    __syncthreads();
    for (int o = 128; o > 0; o >>= 1) {
        if (t < o) { r0[t] += r0[t + o]; r1[t] += r1[t + o]; }
        __syncthreads();
    }
    if (t == 0) {
        part[(c * 8 + n) * 2 + 0] = r0[0];
        part[(c * 8 + n) * 2 + 1] = r1[0];
    }
}

__global__ void finalize_bn2(const float* __restrict__ part,
                             const float* go, const float* bo,
                             float* __restrict__ ab2) {
    int c = threadIdx.x;                 // 0..63
    float s = 0.f, s2 = 0.f;
    for (int n = 0; n < BATCH; ++n) {
        s  += part[(c * 8 + n) * 2 + 0];
        s2 += part[(c * 8 + n) * 2 + 1];
    }
    float mean = s / NTOT;
    float var  = s2 / NTOT - mean * mean;
    float sc = go[c] * rsqrtf(var + BN_EPS);
    ab2[c * 2 + 0] = sc;
    ab2[c * 2 + 1] = bo[c] - mean * sc;
}

// ---------------------------------------------------------------------------
// K7: recompute attention (bit-identical to K6) and write normalized output.
// ---------------------------------------------------------------------------
__global__ __launch_bounds__(256) void write_out(
        const __bf16* __restrict__ yv, const float* __restrict__ ab,
        const float* __restrict__ attn_s, const float* __restrict__ attn_c,
        const float* __restrict__ ab2, float* __restrict__ out) {
    int c = blockIdx.x, n = blockIdx.y, t = threadIdx.x;
    float av = ab[256 + c * 2], bv = ab[256 + c * 2 + 1];
    float ac = attn_c[n * 64 + c];
    float a2 = ab2[c * 2], b2 = ab2[c * 2 + 1];
    int base = (n * 64 + c) * HWSZ, sbase = n * HWSZ;
    for (int i = t; i < HWSZ; i += 256) {
        float v   = sigm(av * (float)yv[base + i] + bv);
        float att = 0.5f * (attn_s[sbase + i] + ac) * v;
        out[base + i] = a2 * att + b2;
    }
}

extern "C" void kernel_launch(void* const* d_in, const int* in_sizes, int n_in,
                              void* d_out, int out_size, void* d_ws, size_t ws_size,
                              hipStream_t stream) {
    (void)in_sizes; (void)n_in; (void)out_size; (void)ws_size;
    const float* x  = (const float*)d_in[0];
    const float* wq = (const float*)d_in[1];
    const float* gq = (const float*)d_in[2];
    const float* bq = (const float*)d_in[3];
    const float* wk = (const float*)d_in[4];
    const float* gk = (const float*)d_in[5];
    const float* bk = (const float*)d_in[6];
    const float* wv = (const float*)d_in[7];
    const float* gv = (const float*)d_in[8];
    const float* bv = (const float*)d_in[9];
    const float* go = (const float*)d_in[10];
    const float* bo = (const float*)d_in[11];
    float* out = (float*)d_out;

    char* ws = (char*)d_ws;
    const size_t YSZ = (size_t)BATCH * CHN * HWSZ * sizeof(__bf16); // 64 MiB
    __bf16* yq = (__bf16*)(ws);
    __bf16* yk = (__bf16*)(ws + YSZ);
    __bf16* yv = (__bf16*)(ws + 2 * YSZ);
    __bf16* pB = (__bf16*)(ws + 3 * YSZ);                 // 110592 bf16
    float* attn_s = (float*)(ws + 3 * YSZ + 262144);      // 524288 f32
    float* attn_c = attn_s + 524288;                      // 512 f32
    float* stats  = attn_c + 512;                         // 384 f32
    float* ab     = stats + 512;                          // 384 f32
    float* part   = ab + 512;                             // 1024 f32
    float* ab2    = part + 1024;                          // 128 f32

    pack_weights<<<432, 256, 0, stream>>>(wq, wk, wv, pB);
    conv_qkv_wmma<<<dim3(4, 256, 8), 128, 0, stream>>>(x, pB, yq, yk, yv);
    bn1_stats<<<dim3(64, 3), 256, 0, stream>>>(yq, yk, yv, stats);
    finalize_bn1<<<1, 192, 0, stream>>>(stats, gq, bq, gk, bk, gv, bv, ab);
    pixel_attn<<<2048, 256, 0, stream>>>(yq, yk, ab, attn_s);
    channel_attn<<<dim3(64, 8), 256, 0, stream>>>(yq, yk, ab, attn_c);
    bn2_partials<<<dim3(64, 8), 256, 0, stream>>>(yv, ab, attn_s, attn_c, part);
    finalize_bn2<<<1, 64, 0, stream>>>(part, go, bo, ab2);
    write_out<<<dim3(64, 8), 256, 0, stream>>>(yv, ab, attn_s, attn_c, ab2, out);
}
